// GPT2Block_41420664602949
// MI455X (gfx1250) — matrix operand order repaired and checked by reading, not compiled, not run on hardware
//
#include <hip/hip_runtime.h>
#include <hip/hip_bf16.h>

typedef __attribute__((ext_vector_type(16))) _Float16 v16h;
typedef __attribute__((ext_vector_type(8)))  float    v8f;

#define Bc 2
#define Sc 2048
#define Dc 1024
#define Hc 16
#define DHc 64
#define DFFc 4096
#define Mc (Bc*Sc)          // 4096 rows

__device__ inline v8f zero8() {
    v8f z = {0.f,0.f,0.f,0.f,0.f,0.f,0.f,0.f};
    return z;
}

__device__ inline v8f wmma_f16(v16h a, v16h b, v8f c) {
    return __builtin_amdgcn_wmma_f32_16x16x32_f16(
        /*neg_a=*/false, a, /*neg_b=*/false, b,
        /*c_mod=*/(short)0, c, /*reuse_a=*/false, /*reuse_b=*/false);
}

// Two contiguous 16-byte chunks -> one 32-byte WMMA operand.
__device__ inline v16h load16(const _Float16* p0, const _Float16* p1) {
    union { v16h v; uint4 u[2]; } r;
    r.u[0] = *(const uint4*)p0;
    r.u[1] = *(const uint4*)p1;
    return r.v;
}

// A operand 16x32 (f16) from row-major [*, ld] at (row 0..15, k 0..31).
// Layout: lanes 0-15: K 0-15 ; lanes 16-31: K 16-31.
__device__ inline v16h load_a16(const _Float16* base, int ld, int lane) {
    int row = lane & 15;
    int k0  = (lane < 16) ? 0 : 16;
    const _Float16* p = base + (size_t)row * ld + k0;
    return load16(p, p + 8);
}

// B operand 32x16 (K x N) where column n is contiguous in memory (row n of a
// transposed matrix). lanes 0-15: K 0-15 ; lanes 16-31: K 16-31.
__device__ inline v16h load_bt16(const _Float16* base, int ld, int lane) {
    int col = lane & 15;
    int k0  = (lane < 16) ? 0 : 16;
    const _Float16* p = base + (size_t)col * ld + k0;
    return load16(p, p + 8);
}

// Wave computes a 32x64 tile. Ping-pong double buffering (no register
// rotation copies): loadY(kb+32) ; wmmaX ; loadX(kb+64) ; wmmaY.
// Requires K % 64 == 0 (true for K = 1024 and 4096 here).
__device__ inline void gemm32x64(const _Float16* A, const _Float16* BT, int K,
                                 int mb, int n0, int lane, v8f c[8]) {
    const _Float16* ar0 = A + (size_t)mb * K;
    const _Float16* ar1 = A + (size_t)(mb + 16) * K;
    const _Float16* bt0 = BT + (size_t)n0 * K;
    const _Float16* bt1 = BT + (size_t)(n0 + 16) * K;
    const _Float16* bt2 = BT + (size_t)(n0 + 32) * K;
    const _Float16* bt3 = BT + (size_t)(n0 + 48) * K;

    v16h aX0 = load_a16(ar0, K, lane);
    v16h aX1 = load_a16(ar1, K, lane);
    v16h bX0 = load_bt16(bt0, K, lane);
    v16h bX1 = load_bt16(bt1, K, lane);
    v16h bX2 = load_bt16(bt2, K, lane);
    v16h bX3 = load_bt16(bt3, K, lane);
    v16h aY0, aY1, bY0, bY1, bY2, bY3;

    int kb = 0;
    for (; kb < K - 64; kb += 64) {
        aY0 = load_a16(ar0 + kb + 32, K, lane);
        aY1 = load_a16(ar1 + kb + 32, K, lane);
        bY0 = load_bt16(bt0 + kb + 32, K, lane);
        bY1 = load_bt16(bt1 + kb + 32, K, lane);
        bY2 = load_bt16(bt2 + kb + 32, K, lane);
        bY3 = load_bt16(bt3 + kb + 32, K, lane);
        c[0] = wmma_f16(aX0, bX0, c[0]); c[4] = wmma_f16(aX1, bX0, c[4]);
        c[1] = wmma_f16(aX0, bX1, c[1]); c[5] = wmma_f16(aX1, bX1, c[5]);
        c[2] = wmma_f16(aX0, bX2, c[2]); c[6] = wmma_f16(aX1, bX2, c[6]);
        c[3] = wmma_f16(aX0, bX3, c[3]); c[7] = wmma_f16(aX1, bX3, c[7]);
        aX0 = load_a16(ar0 + kb + 64, K, lane);
        aX1 = load_a16(ar1 + kb + 64, K, lane);
        bX0 = load_bt16(bt0 + kb + 64, K, lane);
        bX1 = load_bt16(bt1 + kb + 64, K, lane);
        bX2 = load_bt16(bt2 + kb + 64, K, lane);
        bX3 = load_bt16(bt3 + kb + 64, K, lane);
        c[0] = wmma_f16(aY0, bY0, c[0]); c[4] = wmma_f16(aY1, bY0, c[4]);
        c[1] = wmma_f16(aY0, bY1, c[1]); c[5] = wmma_f16(aY1, bY1, c[5]);
        c[2] = wmma_f16(aY0, bY2, c[2]); c[6] = wmma_f16(aY1, bY2, c[6]);
        c[3] = wmma_f16(aY0, bY3, c[3]); c[7] = wmma_f16(aY1, bY3, c[7]);
    }
    // peeled last 64
    aY0 = load_a16(ar0 + kb + 32, K, lane);
    aY1 = load_a16(ar1 + kb + 32, K, lane);
    bY0 = load_bt16(bt0 + kb + 32, K, lane);
    bY1 = load_bt16(bt1 + kb + 32, K, lane);
    bY2 = load_bt16(bt2 + kb + 32, K, lane);
    bY3 = load_bt16(bt3 + kb + 32, K, lane);
    c[0] = wmma_f16(aX0, bX0, c[0]); c[4] = wmma_f16(aX1, bX0, c[4]);
    c[1] = wmma_f16(aX0, bX1, c[1]); c[5] = wmma_f16(aX1, bX1, c[5]);
    c[2] = wmma_f16(aX0, bX2, c[2]); c[6] = wmma_f16(aX1, bX2, c[6]);
    c[3] = wmma_f16(aX0, bX3, c[3]); c[7] = wmma_f16(aX1, bX3, c[7]);
    c[0] = wmma_f16(aY0, bY0, c[0]); c[4] = wmma_f16(aY1, bY0, c[4]);
    c[1] = wmma_f16(aY0, bY1, c[1]); c[5] = wmma_f16(aY1, bY1, c[5]);
    c[2] = wmma_f16(aY0, bY2, c[2]); c[6] = wmma_f16(aY1, bY2, c[6]);
    c[3] = wmma_f16(aY0, bY3, c[3]); c[7] = wmma_f16(aY1, bY3, c[7]);
}

// ---------------- weight convert: fp32 W[K,N] -> f16 WT[N,K] ----------------
__global__ void k_convert_T(const float* __restrict__ W, _Float16* __restrict__ WT,
                            int K, int N) {
    size_t idx = (size_t)blockIdx.x * blockDim.x + threadIdx.x;
    if (idx >= (size_t)K * N) return;
    int n  = (int)(idx % N);
    int kk = (int)(idx / N);
    WT[(size_t)n * K + kk] = (_Float16)W[idx];
}

// ---------------- layernorm: fp32 [rows, D] -> f16 [rows, D] ----------------
__global__ void k_layernorm(const float* __restrict__ x, const float* __restrict__ w,
                            const float* __restrict__ bv, _Float16* __restrict__ y) {
    int row = blockIdx.x;
    int tid = threadIdx.x;
    const float* xr = x + (size_t)row * Dc;
    float s = 0.f, ss = 0.f;
    for (int i = tid; i < Dc; i += 256) {
        float v = xr[i];
        s += v; ss += v * v;
    }
    __shared__ float sh[256], sh2[256];
    sh[tid] = s; sh2[tid] = ss;
    __syncthreads();
    for (int off = 128; off > 0; off >>= 1) {
        if (tid < off) { sh[tid] += sh[tid + off]; sh2[tid] += sh2[tid + off]; }
        __syncthreads();
    }
    float mean = sh[0] * (1.0f / Dc);
    float var  = sh2[0] * (1.0f / Dc) - mean * mean;
    float rstd = rsqrtf(var + 1e-6f);
    _Float16* yr = y + (size_t)row * Dc;
    for (int i = tid; i < Dc; i += 256)
        yr[i] = (_Float16)(w[i] * ((xr[i] - mean) * rstd) + bv[i]);
}

// ---------------- QKV GEMM: x_h[4096,1024] @ WT_attn -> q,k (scaled), vT ----
__global__ void __launch_bounds__(256, 1)
k_gemm_qkv(const _Float16* __restrict__ X, const _Float16* __restrict__ WT,
           const float* __restrict__ bias,
           _Float16* __restrict__ qh, _Float16* __restrict__ kh,
           _Float16* __restrict__ vT) {
    int wid  = (int)((blockIdx.x * blockDim.x + threadIdx.x) >> 5);
    int lane = threadIdx.x & 31;
    const int NT = (3 * Dc) / 64;                 // 48
    int nt = wid % NT, mt = wid / NT;             // mt in [0,128)
    int mb = mt * 32, n0 = nt * 64;
    v8f c[8] = {zero8(), zero8(), zero8(), zero8(),
                zero8(), zero8(), zero8(), zero8()};
    gemm32x64(X, WT, Dc, mb, n0, lane, c);

    int which = n0 / Dc;                          // 0=q 1=k 2=v (64 | 1024)
    int hh    = (n0 % Dc) / DHc;                  // head (64-wide groups)
    int bidx  = mb / Sc, s = mb % Sc;
    int rowoff = (lane & 16) ? 8 : 0;
    int ncol   = lane & 15;
#pragma unroll
    for (int rt = 0; rt < 2; ++rt)
#pragma unroll
        for (int t = 0; t < 4; ++t)
#pragma unroll
            for (int r = 0; r < 8; ++r) {
                int srow = s + rt * 16 + r + rowoff;
                int dh   = t * 16 + ncol;
                float val = c[rt * 4 + t][r] + bias[n0 + dh];
                size_t hb = ((size_t)bidx * Hc + hh);
                if (which == 0)
                    qh[(hb * Sc + srow) * DHc + dh] = (_Float16)(val * 0.125f);
                else if (which == 1)
                    kh[(hb * Sc + srow) * DHc + dh] = (_Float16)val;
                else
                    vT[(hb * DHc + dh) * Sc + srow] = (_Float16)val;
            }
}

// ---------------- flash attention: one wave per (b,h, 16-query block) -------
__global__ void __launch_bounds__(128, 1)
k_attention(const _Float16* __restrict__ qh, const _Float16* __restrict__ kh,
            const _Float16* __restrict__ vT, _Float16* __restrict__ aout) {
    __shared__ __align__(16) _Float16 plds[4][16 * 32];   // per-wave prob tile
    int lane = threadIdx.x & 31;
    int w    = threadIdx.x >> 5;
    int task = blockIdx.x * 4 + w;                // 4 waves / 128-thread block
    const int QT = Sc / 16;                       // 128
    int qt = task % QT;
    int bh = task / QT;                           // b*H + h, 0..31
    int qb = qt * 16;

    const _Float16* qbase = qh + ((size_t)bh * Sc + qb) * DHc;
    const _Float16* kbase = kh + (size_t)bh * Sc * DHc;
    const _Float16* vbase = vT + (size_t)bh * DHc * Sc;

    v16h a0 = load_a16(qbase, DHc, lane);         // K = 0..31 of DH
    v16h a1 = load_a16(qbase + 32, DHc, lane);    // K = 32..63

    v8f o[4] = {zero8(), zero8(), zero8(), zero8()};
    float mrow[8], lrow[8];
#pragma unroll
    for (int r = 0; r < 8; ++r) { mrow[r] = -1e30f; lrow[r] = 0.f; }

    int rowoff = (lane & 16) ? 8 : 0;
    int ncol   = lane & 15;
    int koff   = (lane < 16) ? 0 : 16;
    _Float16* pw = plds[w];
    int nchunks = qb / 32 + 1;

    // preload K-operands for chunk 0 (score B operands, 2 tiles x {lo,hi})
    v16h kb0lo, kb0hi, kb1lo, kb1hi;
    {
        const _Float16* kr = kbase + (size_t)ncol * DHc + koff;
        kb0lo = load16(kr, kr + 8);
        kb0hi = load16(kr + 32, kr + 40);
        const _Float16* kr2 = kr + 16 * DHc;
        kb1lo = load16(kr2, kr2 + 8);
        kb1hi = load16(kr2 + 32, kr2 + 40);
    }

    for (int j = 0; j < nchunks; ++j) {
        int kb = j * 32;
        // ---- scores: two 16x16 tiles over 32 keys, K-dim = DH = 64 ----
        v8f s0 = zero8(), s1 = zero8();
        s0 = wmma_f16(a0, kb0lo, s0);
        s0 = wmma_f16(a1, kb0hi, s0);
        s1 = wmma_f16(a0, kb1lo, s1);
        s1 = wmma_f16(a1, kb1hi, s1);

        // issue next chunk's K-operand loads early (independent of softmax)
        if (j + 1 < nchunks) {
            const _Float16* kr = kbase + (size_t)(kb + 32 + ncol) * DHc + koff;
            kb0lo = load16(kr, kr + 8);
            kb0hi = load16(kr + 32, kr + 40);
            const _Float16* kr2 = kr + 16 * DHc;
            kb1lo = load16(kr2, kr2 + 8);
            kb1hi = load16(kr2 + 32, kr2 + 40);
        }
        // issue this chunk's V-operand loads early too
        v16h bv[4];
#pragma unroll
        for (int t = 0; t < 4; ++t) {
            const _Float16* vr = vbase + (size_t)(t * 16 + ncol) * Sc + kb + koff;
            bv[t] = load16(vr, vr + 8);
        }

        // ---- causal mask + online softmax (per-row stats) ----
#pragma unroll
        for (int r = 0; r < 8; ++r) {
            int qrow = qb + r + rowoff;
            int kc0 = kb + ncol, kc1 = kc0 + 16;
            float v0 = (kc0 <= qrow) ? s0[r] : -10000.0f;
            float v1 = (kc1 <= qrow) ? s1[r] : -10000.0f;
            float mx = fmaxf(v0, v1);
            mx = fmaxf(mx, __shfl_xor(mx, 1));
            mx = fmaxf(mx, __shfl_xor(mx, 2));
            mx = fmaxf(mx, __shfl_xor(mx, 4));
            mx = fmaxf(mx, __shfl_xor(mx, 8));
            float mnew = fmaxf(mrow[r], mx);
            float sc = __expf(mrow[r] - mnew);
            float p0 = __expf(v0 - mnew);
            float p1 = __expf(v1 - mnew);
            float sum = p0 + p1;
            sum += __shfl_xor(sum, 1);
            sum += __shfl_xor(sum, 2);
            sum += __shfl_xor(sum, 4);
            sum += __shfl_xor(sum, 8);
            lrow[r] = lrow[r] * sc + sum;
            mrow[r] = mnew;
            o[0][r] *= sc; o[1][r] *= sc; o[2][r] *= sc; o[3][r] *= sc;
            int prow = r + rowoff;
            pw[prow * 32 + ncol]      = (_Float16)p0;
            pw[prow * 32 + ncol + 16] = (_Float16)p1;
        }
        asm volatile("s_wait_dscnt 0" ::: "memory");
        // ---- P (16x32) @ V (32x64): one WMMA per 16-wide dh tile ----
        v16h ap = load_a16(pw, 32, lane);
#pragma unroll
        for (int t = 0; t < 4; ++t)
            o[t] = wmma_f16(ap, bv[t], o[t]);
    }
    // ---- normalize + merge heads -> attn_out f16 [B,S,D] ----
    int b = bh >> 4, h = bh & 15;
#pragma unroll
    for (int t = 0; t < 4; ++t)
#pragma unroll
        for (int r = 0; r < 8; ++r) {
            int srow = qb + r + rowoff;
            int dcol = h * DHc + t * 16 + ncol;
            aout[((size_t)b * Sc + srow) * Dc + dcol] = (_Float16)(o[t][r] / lrow[r]);
        }
}

// ---------------- GEMM + bias + fp32 residual -> fp32 out -------------------
__global__ void __launch_bounds__(256, 1)
k_gemm_bias_res(const _Float16* __restrict__ A, const _Float16* __restrict__ BT,
                const float* __restrict__ bias, const float* __restrict__ res,
                float* __restrict__ out, int N, int K) {
    int wid  = (int)((blockIdx.x * blockDim.x + threadIdx.x) >> 5);
    int lane = threadIdx.x & 31;
    int NT = N / 64;
    int nt = wid % NT, mt = wid / NT;
    int mb = mt * 32, n0 = nt * 64;
    v8f c[8] = {zero8(), zero8(), zero8(), zero8(),
                zero8(), zero8(), zero8(), zero8()};
    gemm32x64(A, BT, K, mb, n0, lane, c);
    int rowoff = (lane & 16) ? 8 : 0;
    int ncol   = lane & 15;
#pragma unroll
    for (int rt = 0; rt < 2; ++rt)
#pragma unroll
        for (int t = 0; t < 4; ++t)
#pragma unroll
            for (int r = 0; r < 8; ++r) {
                int row = mb + rt * 16 + r + rowoff;
                int col = n0 + t * 16 + ncol;
                size_t idx = (size_t)row * N + col;
                out[idx] = c[rt * 4 + t][r] + bias[col] + res[idx];
            }
}

// ---------------- GEMM + bias + GELU -> f16 out -----------------------------
__global__ void __launch_bounds__(256, 1)
k_gemm_gelu(const _Float16* __restrict__ A, const _Float16* __restrict__ BT,
            const float* __restrict__ bias, _Float16* __restrict__ out,
            int N, int K) {
    int wid  = (int)((blockIdx.x * blockDim.x + threadIdx.x) >> 5);
    int lane = threadIdx.x & 31;
    int NT = N / 64;
    int nt = wid % NT, mt = wid / NT;
    int mb = mt * 32, n0 = nt * 64;
    v8f c[8] = {zero8(), zero8(), zero8(), zero8(),
                zero8(), zero8(), zero8(), zero8()};
    gemm32x64(A, BT, K, mb, n0, lane, c);
    int rowoff = (lane & 16) ? 8 : 0;
    int ncol   = lane & 15;
#pragma unroll
    for (int rt = 0; rt < 2; ++rt)
#pragma unroll
        for (int t = 0; t < 4; ++t)
#pragma unroll
            for (int r = 0; r < 8; ++r) {
                int row = mb + rt * 16 + r + rowoff;
                int col = n0 + t * 16 + ncol;
                float v = c[rt * 4 + t][r] + bias[col];
                float g = 0.5f * v * (1.0f + tanhf(0.7978845608028654f *
                                                   (v + 0.044715f * v * v * v)));
                out[(size_t)row * N + col] = (_Float16)g;
            }
}

extern "C" void kernel_launch(void* const* d_in, const int* in_sizes, int n_in,
                              void* d_out, int out_size, void* d_ws, size_t ws_size,
                              hipStream_t stream) {
    const float* hidden = (const float*)d_in[0];
    const float* w_attn = (const float*)d_in[1];
    const float* b_attn = (const float*)d_in[2];
    const float* w_proj = (const float*)d_in[3];
    const float* b_proj = (const float*)d_in[4];
    const float* w_fc   = (const float*)d_in[5];
    const float* b_fc   = (const float*)d_in[6];
    const float* w_fcp  = (const float*)d_in[7];
    const float* b_fcp  = (const float*)d_in[8];
    const float* ln1w   = (const float*)d_in[9];
    const float* ln1b   = (const float*)d_in[10];
    const float* ln2w   = (const float*)d_in[11];
    const float* ln2b   = (const float*)d_in[12];
    float* out = (float*)d_out;

    char* p = (char*)d_ws;
    auto alloc = [&](size_t bytes) -> char* {
        char* r = p;
        p += (bytes + 255) & ~(size_t)255;
        return r;
    };
    _Float16* x1h  = (_Float16*)alloc((size_t)Mc * Dc * 2);          // LN1(x)
    _Float16* wtat = (_Float16*)alloc((size_t)3 * Dc * Dc * 2);      // W_attn^T
    _Float16* wtpr = (_Float16*)alloc((size_t)Dc * Dc * 2);          // W_proj^T
    _Float16* wtfc = (_Float16*)alloc((size_t)DFFc * Dc * 2);        // W_fc^T
    _Float16* wtfp = (_Float16*)alloc((size_t)Dc * DFFc * 2);        // W_fcp^T
    _Float16* qhp  = (_Float16*)alloc((size_t)Mc * Dc * 2);          // q (scaled)
    _Float16* khp  = (_Float16*)alloc((size_t)Mc * Dc * 2);          // k
    _Float16* vtp  = (_Float16*)alloc((size_t)Mc * Dc * 2);          // v^T
    _Float16* aoh  = (_Float16*)alloc((size_t)Mc * Dc * 2);          // attn out
    float*    hbuf = (float*)   alloc((size_t)Mc * Dc * 4);          // h fp32
    _Float16* y2h  = (_Float16*)alloc((size_t)Mc * Dc * 2);          // LN2(h)
    _Float16* act  = (_Float16*)alloc((size_t)Mc * DFFc * 2);        // gelu act

    // 1) weights -> f16 transposed
    k_convert_T<<<(3 * Dc * Dc + 255) / 256, 256, 0, stream>>>(w_attn, wtat, Dc, 3 * Dc);
    k_convert_T<<<(Dc * Dc + 255) / 256, 256, 0, stream>>>(w_proj, wtpr, Dc, Dc);
    k_convert_T<<<(Dc * DFFc + 255) / 256, 256, 0, stream>>>(w_fc, wtfc, Dc, DFFc);
    k_convert_T<<<(DFFc * Dc + 255) / 256, 256, 0, stream>>>(w_fcp, wtfp, DFFc, Dc);

    // 2) LN1
    k_layernorm<<<Mc, 256, 0, stream>>>(hidden, ln1w, ln1b, x1h);

    // 3) QKV GEMM: waves = (4096/32) * (3072/64) = 128*48 ; 8 waves/block
    k_gemm_qkv<<<(128 * 48) / 8, 256, 0, stream>>>(x1h, wtat, b_attn, qhp, khp, vtp);

    // 4) flash attention (4096 wave tasks, 4 waves/block)
    k_attention<<<1024, 128, 0, stream>>>(qhp, khp, vtp, aoh);

    // 5) proj + residual -> h fp32   (waves = 128*16)
    k_gemm_bias_res<<<(128 * 16) / 8, 256, 0, stream>>>(aoh, wtpr, b_proj, hidden,
                                                        hbuf, Dc, Dc);
    // 6) LN2
    k_layernorm<<<Mc, 256, 0, stream>>>(hbuf, ln2w, ln2b, y2h);

    // 7) fc + GELU   (waves = 128*64)
    k_gemm_gelu<<<(128 * 64) / 8, 256, 0, stream>>>(y2h, wtfc, b_fc, act, DFFc, Dc);

    // 8) fc_proj + residual -> output fp32   (waves = 128*16)
    k_gemm_bias_res<<<(128 * 16) / 8, 256, 0, stream>>>(act, wtfp, b_fcp, hbuf,
                                                        out, Dc, DFFc);
}
